// MultiHeadSelfAttention_27994596836189
// MI455X (gfx1250) — matrix operand order repaired and checked by reading, not compiled, not run on hardware
//
#include <hip/hip_runtime.h>

// ---------------------------------------------------------------------------
// MHSA forward for MI455X (gfx1250): bf16 WMMA, f32 accumulate, async LDS
// staging + software-pipelined GEMMs. B=2, N=2048, D=1024, H=16, hd=64.
// ---------------------------------------------------------------------------

typedef __attribute__((ext_vector_type(16))) __bf16 bf16x16;
typedef __attribute__((ext_vector_type(8)))  float  f32x8;
typedef __attribute__((ext_vector_type(4)))  unsigned int u32x4;

#define DEV __device__ __forceinline__

#if defined(__HIP_DEVICE_COMPILE__) && __has_builtin(__builtin_amdgcn_global_load_async_to_lds_b128)
#define HAS_ASYNC_LDS 1
#else
#define HAS_ASYNC_LDS 0
#endif

#if defined(__HIP_DEVICE_COMPILE__) && __has_builtin(__builtin_amdgcn_s_wait_asynccnt)
#define WAIT_ASYNC(N) __builtin_amdgcn_s_wait_asynccnt(N)
#elif defined(__HIP_DEVICE_COMPILE__) && HAS_ASYNC_LDS
#define WAIT_ASYNC(N) asm volatile("s_wait_asynccnt %0" ::"n"(N) : "memory")
#else
#define WAIT_ASYNC(N)
#endif

union Frag {
    bf16x16 v;
    u32x4   q[2];
};

// A-matrix fragment (16x32 bf16, MxK). Lane ln (0..15) = row M=ln.
// Lanes 0-15 : K[k0+0..15] contiguous
// Lanes 16-31: K[k0+16..31] contiguous
DEV bf16x16 load_a_frag(const __bf16* rowp, int hi) {
    Frag u;
    u.q[0] = *(const u32x4*)(rowp + hi * 16);
    u.q[1] = *(const u32x4*)(rowp + hi * 16 + 8);
    return u.v;
}

// B-matrix fragment (32x16 bf16, KxN). Lane ln = column; K contiguous,
// lanes 0-15 hold K[k0+0..15], lanes 16-31 hold K[k0+16..31].
DEV bf16x16 load_b_frag(const __bf16* colp) {
    Frag u;
    const u32x4* q = (const u32x4*)colp;
    u.q[0] = q[0];
    u.q[1] = q[1];
    return u.v;
}

DEV f32x8 wmma_bf16(bf16x16 a, bf16x16 b, f32x8 c) {
    return __builtin_amdgcn_wmma_f32_16x16x32_bf16(
        false, a, false, b, (short)0, c, false, false);
}

#if HAS_ASYNC_LDS
typedef int v4i_vs __attribute__((vector_size(16)));
typedef __attribute__((address_space(1))) v4i_vs* g_v4i_ptr;
typedef __attribute__((address_space(3))) v4i_vs* l_v4i_ptr;
#endif

// 16-byte global -> LDS copy; async DMA when the toolchain exposes it.
DEV void cp16_g2l(const __bf16* g, __bf16* l) {
#if HAS_ASYNC_LDS
    void* gg = (void*)const_cast<__bf16*>(g);
    __builtin_amdgcn_global_load_async_to_lds_b128(
        (g_v4i_ptr)(__attribute__((address_space(1))) void*)gg,
        (l_v4i_ptr)(__attribute__((address_space(3))) void*)(void*)l, 0, 0);
#else
    *(u32x4*)l = *(const u32x4*)g;
#endif
}

// ---------------------------------------------------------------------------
// Kernel 1: f32 -> bf16 conversion
// ---------------------------------------------------------------------------
__global__ void cvt_f32_to_bf16(const float* __restrict__ src,
                                __bf16* __restrict__ dst, int n) {
    int i = blockIdx.x * blockDim.x + threadIdx.x;
    int stride = gridDim.x * blockDim.x;
    for (; i < n; i += stride) dst[i] = (__bf16)src[i];
}

// ---------------------------------------------------------------------------
// Software-pipelined 16x64 per-wave GEMM body (K=1024), shared by kernels 2/4.
// ---------------------------------------------------------------------------
DEV void gemm_body(const __bf16* __restrict__ X, const __bf16* __restrict__ W,
                   int m0, int n0, int ln, int hi, f32x8 c[4]) {
    const __bf16* arow = X + (size_t)(m0 + ln) * 1024;
    const __bf16* brow = W + (size_t)(n0 + ln) * 1024 + hi * 16;

    bf16x16 a_c = load_a_frag(arow, hi);
    bf16x16 b_c[4];
#pragma unroll
    for (int j = 0; j < 4; ++j) b_c[j] = load_b_frag(brow + j * 16384);

#pragma unroll 2
    for (int k0 = 0; k0 < 1024 - 32; k0 += 32) {
        // Prefetch next K-slice before consuming the current one.
        bf16x16 a_n = load_a_frag(arow + k0 + 32, hi);
        bf16x16 b_n[4];
#pragma unroll
        for (int j = 0; j < 4; ++j)
            b_n[j] = load_b_frag(brow + j * 16384 + k0 + 32);
#pragma unroll
        for (int j = 0; j < 4; ++j) c[j] = wmma_bf16(a_c, b_c[j], c[j]);
        a_c = a_n;
#pragma unroll
        for (int j = 0; j < 4; ++j) b_c[j] = b_n[j];
    }
#pragma unroll
    for (int j = 0; j < 4; ++j) c[j] = wmma_bf16(a_c, b_c[j], c[j]);
}

// ---------------------------------------------------------------------------
// Kernel 2: QKV projection -> Q[BH,N,64], K[BH,N,64], Vt[BH,64,N] (bf16).
// grid = (48, 64), block = 128.
// ---------------------------------------------------------------------------
__global__ void __launch_bounds__(128)
qkv_gemm(const __bf16* __restrict__ X, const __bf16* __restrict__ W,
         const float* __restrict__ bias,
         __bf16* __restrict__ Qo, __bf16* __restrict__ Ko,
         __bf16* __restrict__ Vt) {
    const int lane = threadIdx.x & 31;
    const int wave = threadIdx.x >> 5;
    const int hi   = (lane >> 4) & 1;
    const int ln   = lane & 15;
    const int m0   = blockIdx.y * 64 + wave * 16;
    const int n0   = blockIdx.x * 64;

    f32x8 c[4] = {};
    gemm_body(X, W, m0, n0, ln, hi, c);

#pragma unroll
    for (int j = 0; j < 4; ++j) {
        const int e     = n0 + j * 16 + ln;   // output feature 0..3071
        const float bv  = bias[e];
        const int which = e >> 10;            // 0=Q 1=K 2=V
        const int rem   = e & 1023;
        const int h     = rem >> 6;
        const int d     = rem & 63;
#pragma unroll
        for (int r = 0; r < 8; ++r) {
            const int row = m0 + r + hi * 8;  // b*2048+n
            const int bb  = row >> 11;
            const int nn  = row & 2047;
            const size_t bh = (size_t)(bb * 16 + h);
            const float val = c[j][r] + bv;
            if (which == 0)      Qo[(bh * 2048 + nn) * 64 + d] = (__bf16)val;
            else if (which == 1) Ko[(bh * 2048 + nn) * 64 + d] = (__bf16)val;
            else                 Vt[(bh * 64 + d) * 2048 + nn] = (__bf16)val;
        }
    }
}

// ---------------------------------------------------------------------------
// Kernel 3: flash attention per (b,h). grid = (32, 32), block = 128 (4 waves).
// K/V tiles (32 keys) staged in LDS via double-buffered async DMA, shared by
// all 4 waves; each wave owns 16 query rows.
// ---------------------------------------------------------------------------
__global__ void __launch_bounds__(128)
attn_fwd(const __bf16* __restrict__ Q, const __bf16* __restrict__ K,
         const __bf16* __restrict__ Vt, __bf16* __restrict__ Ctx) {
    __shared__ __align__(16) __bf16 Ktile[2][32][64];  // [buf][key][feat] 8KB
    __shared__ __align__(16) __bf16 Vtile[2][64][32];  // [buf][hd][key]   8KB
    __shared__ __align__(16) __bf16 Pt[4][16][32];     // per-wave P tile  4KB

    const int tid  = threadIdx.x;
    const int lane = tid & 31;
    const int wave = tid >> 5;
    const int hi   = (lane >> 4) & 1;
    const int ln   = lane & 15;
    const int bh   = blockIdx.y;                 // b*16 + h
    const int q0   = blockIdx.x * 64 + wave * 16;

    const __bf16* Qb = Q  + (size_t)bh * 2048 * 64;
    const __bf16* Kb = K  + (size_t)bh * 2048 * 64;
    const __bf16* Vb = Vt + (size_t)bh * 64 * 2048;

    // Per-thread chunk assignment for tile staging (4 x b128 per thread).
    const int kkey = tid >> 3, koff = (tid & 7) * 8;   // K tile: 32x64 bf16
    const int vhd  = tid >> 2, voff = (tid & 3) * 8;   // V tile: 64x32 bf16

    // Q fragments for hd=64 (two K=32 fragments), kept in VGPRs.
    const __bf16* qrow = Qb + (size_t)(q0 + ln) * 64;
    const bf16x16 aq0 = load_a_frag(qrow, hi);
    const bf16x16 aq1 = load_a_frag(qrow + 32, hi);

    float mrow[8], lrow[8];
    f32x8 o[4] = {};
#pragma unroll
    for (int r = 0; r < 8; ++r) { mrow[r] = -3.0e38f; lrow[r] = 0.0f; }

    const float scale = 0.125f;  // 1/sqrt(64)

    // ---- stage key/value tile kt -> LDS buffer buf (4 async b128/thread)
    auto stage = [&](int kt, int buf) {
        cp16_g2l(Kb + (size_t)(kt + kkey) * 64 + koff,        &Ktile[buf][kkey][koff]);
        cp16_g2l(Kb + (size_t)(kt + kkey + 16) * 64 + koff,   &Ktile[buf][kkey + 16][koff]);
        cp16_g2l(Vb + (size_t)vhd * 2048 + kt + voff,         &Vtile[buf][vhd][voff]);
        cp16_g2l(Vb + (size_t)(vhd + 32) * 2048 + kt + voff,  &Vtile[buf][vhd + 32][voff]);
    };

    stage(0, 0);
    for (int i = 0; i < 64; ++i) {
        const int kt  = i * 32;
        const int buf = i & 1;
        if (i < 63) {                // prefetch next tile, wait for current
            stage(kt + 32, buf ^ 1);
            WAIT_ASYNC(4);
        } else {
            WAIT_ASYNC(0);
        }
        __syncthreads();             // tile[buf] visible to all waves

        // ---- S tile 16x32 = two 16x16 C fragments (from LDS K tile)
        f32x8 s[2];
#pragma unroll
        for (int t = 0; t < 2; ++t) {
            const __bf16* krow = &Ktile[buf][t * 16 + ln][hi * 16];
            f32x8 z = {};
            z = wmma_bf16(aq0, load_b_frag(krow), z);       // features 0..31
            z = wmma_bf16(aq1, load_b_frag(krow + 32), z);  // features 32..63
            s[t] = z;
        }

        // ---- online softmax (rows live per lane-half, 16-lane reductions)
#pragma unroll
        for (int r = 0; r < 8; ++r) {
            float s0 = s[0][r] * scale;
            float s1 = s[1][r] * scale;
            float mx = fmaxf(s0, s1);
            mx = fmaxf(mx, __shfl_xor(mx, 1, 16));
            mx = fmaxf(mx, __shfl_xor(mx, 2, 16));
            mx = fmaxf(mx, __shfl_xor(mx, 4, 16));
            mx = fmaxf(mx, __shfl_xor(mx, 8, 16));
            const float mnew  = fmaxf(mrow[r], mx);
            const float alpha = __expf(mrow[r] - mnew);
            const float p0 = __expf(s0 - mnew);
            const float p1 = __expf(s1 - mnew);
            float sum = p0 + p1;
            sum += __shfl_xor(sum, 1, 16);
            sum += __shfl_xor(sum, 2, 16);
            sum += __shfl_xor(sum, 4, 16);
            sum += __shfl_xor(sum, 8, 16);
            lrow[r] = lrow[r] * alpha + sum;
            mrow[r] = mnew;
#pragma unroll
            for (int j = 0; j < 4; ++j) o[j][r] *= alpha;

            const int row = r + hi * 8;
            Pt[wave][row][ln]      = (__bf16)p0;
            Pt[wave][row][16 + ln] = (__bf16)p1;
        }
        __syncthreads();
        // Reload P as an A fragment (16x32, K = key index).
        const bf16x16 pa = load_a_frag(&Pt[wave][ln][0], hi);

        // ---- O += P x V (from LDS V tile; rows contiguous in keys)
#pragma unroll
        for (int j = 0; j < 4; ++j) {
            const __bf16* vrow = &Vtile[buf][j * 16 + ln][hi * 16];
            o[j] = wmma_bf16(pa, load_b_frag(vrow), o[j]);
        }
        __syncthreads();             // done with buf before it is re-staged
    }

    // ---- epilogue: normalize, write Ctx[b*2048+n][h*64+d] (bf16)
    const int h  = bh & 15;
    const int bb = bh >> 4;
#pragma unroll
    for (int r = 0; r < 8; ++r) {
        const float inv = 1.0f / lrow[r];
        const int row = bb * 2048 + q0 + r + hi * 8;
#pragma unroll
        for (int j = 0; j < 4; ++j) {
            const int col = h * 64 + j * 16 + ln;
            Ctx[(size_t)row * 1024 + col] = (__bf16)(o[j][r] * inv);
        }
    }
}

// ---------------------------------------------------------------------------
// Kernel 4: output projection -> f32 d_out. grid = (16, 64), block = 128.
// ---------------------------------------------------------------------------
__global__ void __launch_bounds__(128)
out_gemm(const __bf16* __restrict__ X, const __bf16* __restrict__ W,
         const float* __restrict__ bias, float* __restrict__ out) {
    const int lane = threadIdx.x & 31;
    const int wave = threadIdx.x >> 5;
    const int hi   = (lane >> 4) & 1;
    const int ln   = lane & 15;
    const int m0   = blockIdx.y * 64 + wave * 16;
    const int n0   = blockIdx.x * 64;

    f32x8 c[4] = {};
    gemm_body(X, W, m0, n0, ln, hi, c);

#pragma unroll
    for (int j = 0; j < 4; ++j) {
        const int col  = n0 + j * 16 + ln;
        const float bv = bias[col];
#pragma unroll
        for (int r = 0; r < 8; ++r) {
            const int row = m0 + r + hi * 8;
            out[(size_t)row * 1024 + col] = c[j][r] + bv;
        }
    }
}

// ---------------------------------------------------------------------------
// Workspace layout (bytes), total 48 MB:
//   xb@0 (8M) | wqkv@8M (6M) | wout@14M (2M) | Q@16M (8M) | K@24M (8M)
//   Vt@32M (8M) | Ctx@40M (8M)
// ---------------------------------------------------------------------------
extern "C" void kernel_launch(void* const* d_in, const int* in_sizes, int n_in,
                              void* d_out, int out_size, void* d_ws, size_t ws_size,
                              hipStream_t stream) {
    (void)in_sizes; (void)n_in; (void)out_size; (void)ws_size;

    const float* x    = (const float*)d_in[0];
    const float* Wqkv = (const float*)d_in[1];
    const float* bqkv = (const float*)d_in[2];
    const float* Wout = (const float*)d_in[3];
    const float* bout = (const float*)d_in[4];

    char* ws = (char*)d_ws;
    __bf16* xb  = (__bf16*)(ws + 0);
    __bf16* wqb = (__bf16*)(ws + 8388608);
    __bf16* wob = (__bf16*)(ws + 14680064);
    __bf16* Qb  = (__bf16*)(ws + 16777216);
    __bf16* Kb  = (__bf16*)(ws + 25165824);
    __bf16* Vtb = (__bf16*)(ws + 33554432);
    __bf16* Ctx = (__bf16*)(ws + 41943040);

    cvt_f32_to_bf16<<<2048, 256, 0, stream>>>(x,    xb,  4096 * 1024);
    cvt_f32_to_bf16<<<2048, 256, 0, stream>>>(Wqkv, wqb, 3072 * 1024);
    cvt_f32_to_bf16<<<1024, 256, 0, stream>>>(Wout, wob, 1024 * 1024);

    dim3 blk(128);
    qkv_gemm<<<dim3(48, 64), blk, 0, stream>>>(xb, wqb, bqkv, Qb, Kb, Vtb);
    attn_fwd<<<dim3(32, 32), blk, 0, stream>>>(Qb, Kb, Vtb, Ctx);
    out_gemm<<<dim3(16, 64), blk, 0, stream>>>(Ctx, wob, bout, (float*)d_out);
}
